// MACRO_SHARED_NOENC_63694364999895
// MI455X (gfx1250) — matrix operand order repaired and checked by reading, not compiled, not run on hardware
//
#include <hip/hip_runtime.h>
#include <cstdint>
#include <cstddef>

// ---------------------------------------------------------------------------
// Model constants (from reference)
// ---------------------------------------------------------------------------
constexpr int kT = 64, kB = 256, kA = 5;
constexpr int kX = 2, kY = 10, kZ = 64, kM = 64, kH = 512;
constexpr int kL = 2;
constexpr int kAB = kA * kB;
constexpr int K_DM1 = 608;   // 10+64+512 = 586 -> pad to 19*32
constexpr int K_D1  = 928;   // 10+320+64+512 = 906 -> pad to 29*32
constexpr int K_GMI = 96;    // 2+64 = 66 -> pad to 3*32
constexpr int K_GMA = 640;   // 320+320 (already 20*32)
constexpr int N_X   = 16;    // X_DIM=2 head padded to one 16-wide tile

typedef __attribute__((ext_vector_type(8)))  float  v8f;
typedef __attribute__((ext_vector_type(16))) __bf16 v16bf;

// ---------------------------------------------------------------------------
// Device helpers
// ---------------------------------------------------------------------------
__device__ __forceinline__ unsigned short f2bf(float f) {
  unsigned u = __float_as_uint(f);
  u += 0x7FFFu + ((u >> 16) & 1u);          // round-to-nearest-even
  return (unsigned short)(u >> 16);
}

__device__ __forceinline__ unsigned pcg_next(unsigned &state) {
  unsigned x = state;
  state = state * 747796405u + 2891336453u;
  x = ((x >> ((x >> 28u) + 4u)) ^ x) * 277803737u;
  return (x >> 22u) ^ x;
}

__device__ __forceinline__ float rnormf(unsigned seed, unsigned idx) {
  unsigned s = (seed * 2654435761u) ^ (idx * 0x9E3779B9u) ^ 0xA511E9B3u;
  unsigned a = pcg_next(s), b = pcg_next(s);
  float u1 = ((a >> 8) + 1u) * (1.0f / 16777216.0f);
  float u2 = (b >> 8) * (1.0f / 16777216.0f);
  return sqrtf(-2.0f * logf(u1)) * cosf(6.28318530718f * u2);
}

__device__ __forceinline__ float softplusf(float x) {
  return (x > 20.f) ? x : log1pf(expf(x));
}

// ---------------------------------------------------------------------------
// WMMA GEMM: Out[b, m, n] = act( sum_k In[b, m, k] * W[b, n, k] + bias[b, n] )
// In: bf16 (M x Kpad) row-major. W: bf16 (N x Kpad) row-major. Out: f32 (M x N).
//
// Block = 256 threads = 8 waves, tile = 64 rows x 128 cols.
// Per K-step (32): the block stages a 64x32 bf16 A-tile into LDS with
// GLOBAL_LOAD_ASYNC_TO_LDS_B128 (ASYNCcnt path), each wave holds one 32x16
// B fragment and issues 4 WMMAs against the 4 A sub-tiles (4x B reuse).
// ---------------------------------------------------------------------------
__global__ void k_gemm(const unsigned short* __restrict__ In, size_t inBS,
                       const unsigned short* __restrict__ W, size_t wBS,
                       const float* __restrict__ bias, size_t bBS, int Nbias,
                       float* __restrict__ Out, size_t oBS,
                       int M, int N, int Kpad, int act)
{
  const int batch = blockIdx.z;
  const unsigned short* inB = In + (size_t)batch * inBS;
  const unsigned short* wB  = W  + (size_t)batch * wBS;
  const float* bb           = bias + (size_t)batch * bBS;
  float* outB               = Out + (size_t)batch * oBS;

  const int mBase = blockIdx.y * 64;
  const int wave  = threadIdx.x >> 5;
  const int lane  = threadIdx.x & 31;
  const int nTile = blockIdx.x * 8 + wave;
  const bool active = (nTile * 16) < N;

  __shared__ unsigned short As[64 * 32];

  const v8f vzero = {0.f, 0.f, 0.f, 0.f, 0.f, 0.f, 0.f, 0.f};
  v8f acc[4] = {vzero, vzero, vzero, vzero};

  // B fragment: lane n (0..15) -> column n of tile, K=0..15 ; lane n+16 -> K=16..31
  const int colW = nTile * 16 + (lane & 15);
  const unsigned short* wrow =
      wB + (size_t)(active ? colW : 0) * Kpad + ((lane >= 16) ? 16 : 0);

  // A fragment: lanes 0..15 hold rows 0..15, K=0..15; lanes 16..31 K=16..31
  const int arow  = lane & 15;
  const int koffA = (lane >= 16) ? 16 : 0;

  // async A-tile staging: 256 threads x 16B = 64 rows x 32 cols bf16 (4 KB)
  const int tid = threadIdx.x;
  const int sr  = tid >> 2;            // 0..63 (row within block tile)
  const int sc  = (tid & 3) * 8;       // half-offset 0/8/16/24
  // LDS byte address: generic pointer low 32 bits == LDS address (ISA 10.2)
  const unsigned ldsAddr = (unsigned)(uintptr_t)(&As[sr * 32 + sc]);
  const unsigned gBase =
      ((unsigned)(mBase + sr) * (unsigned)Kpad + (unsigned)sc) * 2u;

  for (int k0 = 0; k0 < Kpad; k0 += 32) {
    unsigned gOff = gBase + (unsigned)k0 * 2u;
    // memory -> LDS, tracked by ASYNCcnt (GVS addressing: saddr base + vgpr off)
    asm volatile("global_load_async_to_lds_b128 %0, %1, %2"
                 :
                 : "v"(ldsAddr), "v"(gOff), "s"(inB)
                 : "memory");
    asm volatile("s_wait_asynccnt 0" ::: "memory");
    __syncthreads();
    if (k0 + 32 < Kpad) __builtin_prefetch(wrow + k0 + 32, 0, 0);
    if (active) {
      union { uint4 q[2]; v16bf v; } bfr;
      bfr.q[0] = *(const uint4*)(wrow + k0);
      bfr.q[1] = *(const uint4*)(wrow + k0 + 8);
      #pragma unroll
      for (int m = 0; m < 4; ++m) {
        union { unsigned short s[16]; v16bf v; } af;
        #pragma unroll
        for (int i = 0; i < 8; ++i) {
          int kk = 2 * i + koffA;
          af.s[2 * i]     = As[(m * 16 + arow) * 32 + kk];
          af.s[2 * i + 1] = As[(m * 16 + arow) * 32 + kk + 1];
        }
        acc[m] = __builtin_amdgcn_wmma_f32_16x16x32_bf16(
            false, af.v, false, bfr.v, (short)0, acc[m], false, false);
      }
    }
    __syncthreads();
  }

  if (active) {
    float bv = (colW < Nbias) ? bb[colW] : 0.f;
    #pragma unroll
    for (int m = 0; m < 4; ++m) {
      #pragma unroll
      for (int r = 0; r < 8; ++r) {
        int row = mBase + m * 16 + r + ((lane >= 16) ? 8 : 0);
        float v = acc[m][r] + bv;
        if (act == 1) v = fmaxf(v, 0.f);
        outB[(size_t)row * N + colW] = v;
      }
    }
  }
}

// ---------------------------------------------------------------------------
// f32 -> bf16 with zero padding (weights or activations).
// grid: (ceil(Kpad/256), rowsPad, batch)
// ---------------------------------------------------------------------------
__global__ void k_pack_pad_bf16(const float* __restrict__ src, size_t sBS,
                                unsigned short* __restrict__ dst, size_t dBS,
                                int rows, int K, int Kpad)
{
  int k = blockIdx.x * 256 + threadIdx.x;
  if (k >= Kpad) return;
  int r = blockIdx.y;
  int batch = blockIdx.z;
  float v = (r < rows && k < K) ? src[(size_t)batch * sBS + (size_t)r * K + k] : 0.f;
  dst[(size_t)batch * dBS + (size_t)r * Kpad + k] = f2bf(v);
}

// ---------------------------------------------------------------------------
// Elementwise kernels
// ---------------------------------------------------------------------------
__global__ void k_zero(float* p, long n) {
  long i = (long)blockIdx.x * 256 + threadIdx.x;
  if (i < n) p[i] = 0.f;
}

__global__ void k_copy2(const float* __restrict__ s, float* d1, float* d2, int n) {
  int i = blockIdx.x * 256 + threadIdx.x;
  if (i < n) { d1[i] = s[i]; d2[i] = s[i]; }
}

// out = mu + softplus(sdraw) * N(0,1)
__global__ void k_sample_gauss(const float* __restrict__ mu, const float* __restrict__ sdr,
                               float* __restrict__ out, long n, unsigned seed)
{
  long i = (long)blockIdx.x * 256 + threadIdx.x;
  if (i >= n) return;
  out[i] = mu[i] + softplusf(sdr[i]) * rnormf(seed, (unsigned)i);
}

// GRU cell: h = (1-z)*tanh(nx + r*nh) + z*h ; r,z = sigmoid of summed gates
__global__ void k_gru_cell(const float* __restrict__ gx, const float* __restrict__ gh,
                           size_t gBS, float* __restrict__ h, size_t hBS,
                           unsigned short* __restrict__ hbf, int Bc, int Hc)
{
  int batch = blockIdx.z;
  int i = blockIdx.x * 256 + threadIdx.x;
  if (i >= Bc * Hc) return;
  int b = i / Hc, j = i % Hc;
  const float* gxb = gx + (size_t)batch * gBS;
  const float* ghb = gh + (size_t)batch * gBS;
  float* hb = h + (size_t)batch * hBS;
  size_t o = (size_t)b * 3 * Hc;
  float r = 1.f / (1.f + expf(-(gxb[o + j] + ghb[o + j])));
  float z = 1.f / (1.f + expf(-(gxb[o + Hc + j] + ghb[o + Hc + j])));
  float nv = tanhf(gxb[o + 2 * Hc + j] + r * ghb[o + 2 * Hc + j]);
  float hv = (1.f - z) * nv + z * hb[(size_t)b * Hc + j];
  hb[(size_t)b * Hc + j] = hv;
  if (hbf) hbf[(size_t)batch * Bc * Hc + (size_t)b * Hc + j] = f2bf(hv);
}

// dm1 input: [y_b(10) | z_ma(64) | hma(512)] -> bf16 (A,B,608)
__global__ void k_concat_dm1(const float* __restrict__ yb, const float* __restrict__ zma,
                             const float* __restrict__ hma, unsigned short* __restrict__ dst)
{
  long i = (long)blockIdx.x * 256 + threadIdx.x;
  if (i >= (long)kAB * K_DM1) return;
  int a = (int)(i / ((long)kB * K_DM1));
  int rem = (int)(i % ((long)kB * K_DM1));
  int b = rem / K_DM1, k = rem % K_DM1;
  float v = 0.f;
  if (k < kY)            v = yb[b * kY + k];
  else if (k < kY + kZ)  v = zma[((size_t)a * kB + b) * kZ + (k - kY)];
  else if (k < 586)      v = hma[(size_t)b * kH + (k - kY - kZ)];
  dst[i] = f2bf(v);
}

// gru_macro input: [m_cat(320) | z_cat(320)] -> bf16 (B,640)
__global__ void k_concat_gmacro(const float* __restrict__ m_t, const float* __restrict__ zma,
                                unsigned short* __restrict__ dst)
{
  int i = blockIdx.x * 256 + threadIdx.x;
  if (i >= kB * K_GMA) return;
  int b = i / K_GMA, k = i % K_GMA;
  float v;
  if (k < kA * kM) {
    int a = k >> 6, j = k & 63;
    v = m_t[((size_t)a * kB + b) * kM + j];
  } else {
    int kk = k - kA * kM;
    int a = kk >> 6, j = kk & 63;
    v = zma[((size_t)a * kB + b) * kZ + j];
  }
  dst[i] = f2bf(v);
}

// d1 input: [y_b(10) | m_b(320) | z(64) | hmi(512)] -> bf16 (A,B,928)
__global__ void k_concat_d1(const float* __restrict__ yb, const float* __restrict__ m_t,
                            const float* __restrict__ zmi, const float* __restrict__ h_mi,
                            unsigned short* __restrict__ dst)
{
  long i = (long)blockIdx.x * 256 + threadIdx.x;
  if (i >= (long)kAB * K_D1) return;
  int a = (int)(i / ((long)kB * K_D1));
  int rem = (int)(i % ((long)kB * K_D1));
  int b = rem / K_D1, k = rem % K_D1;
  float v = 0.f;
  if (k < kY) v = yb[b * kY + k];
  else if (k < kY + kA * kM) {
    int j2 = k - kY;
    v = m_t[((size_t)(j2 >> 6) * kB + b) * kM + (j2 & 63)];
  } else if (k < kY + kA * kM + kZ) {
    v = zmi[((size_t)a * kB + b) * kZ + (k - kY - kA * kM)];
  } else if (k < 906) {
    // h_mi layout (A, L, B, H), slice l = L-1
    v = h_mi[(((size_t)a * kL + (kL - 1)) * kB + b) * kH + (k - kY - kA * kM - kZ)];
  }
  dst[i] = f2bf(v);
}

// gru_micro input: [x_new(2) | z(64)] -> bf16 (A,B,96)   (x_new stored ld=16)
__global__ void k_concat_gmicro(const float* __restrict__ xnew, const float* __restrict__ zmi,
                                unsigned short* __restrict__ dst)
{
  long i = (long)blockIdx.x * 256 + threadIdx.x;
  if (i >= (long)kAB * K_GMI) return;
  int a = (int)(i / ((long)kB * K_GMI));
  int rem = (int)(i % ((long)kB * K_GMI));
  int b = rem / K_GMI, k = rem % K_GMI;
  float v = 0.f;
  if (k < kX)            v = xnew[((size_t)a * kB + b) * N_X + k];
  else if (k < kX + kZ)  v = zmi[((size_t)a * kB + b) * kZ + (k - kX)];
  dst[i] = f2bf(v);
}

// Macro categorical sample (Gumbel-max == categorical(log_softmax(logits))),
// with given-goal override for batch 0. Writes one-hot m_t and goals output.
__global__ void k_macro_sample(const float* __restrict__ logits,
                               const float* __restrict__ goals_t,  // (B,A) at step t
                               float* __restrict__ m_t,
                               float* __restrict__ goals_out,
                               float* __restrict__ goals_out2,
                               unsigned seed)
{
  int i = blockIdx.x * 256 + threadIdx.x;
  if (i >= kAB) return;
  int a = i / kB, b = i % kB;
  float* mrow = m_t + (size_t)i * kM;
  float g0 = goals_t[0 * kA + a];
  int sel = 0;
  if (g0 >= 0.f) {
    #pragma unroll 8
    for (int j = 0; j < kM; ++j) mrow[j] = 0.f;
    if (b == 0) {
      sel = (int)g0;
      if (sel < 0) sel = 0;
      if (sel >= kM) sel = kM - 1;
      mrow[sel] = 1.f;
    }
  } else {
    const float* lg = logits + (size_t)i * kM;
    unsigned s = (seed * 0x85EBCA6Bu) ^ (unsigned)i * 0xC2B2AE35u;
    float best = -1e30f;
    int bi = 0;
    for (int j = 0; j < kM; ++j) {
      unsigned hrnd = pcg_next(s);
      float u = ((hrnd >> 8) + 1u) * (1.0f / 16777216.0f);
      float g = lg[j] - logf(-logf(u));
      if (g > best) { best = g; bi = j; }
    }
    #pragma unroll 8
    for (int j = 0; j < kM; ++j) mrow[j] = 0.f;
    mrow[bi] = 1.f;
    sel = bi;
  }
  goals_out[b * kA + a] = (float)sel;
  if (goals_out2) goals_out2[b * kA + a] = (float)sel;
}

// y_next[b, :] = x_new transposed (A,B,2)->(B,10); update carry + write ret row
__global__ void k_ynext(const float* __restrict__ xnew, float* __restrict__ yb,
                        float* __restrict__ ret)
{
  int i = blockIdx.x * 256 + threadIdx.x;
  if (i >= kB * kY) return;
  int b = i / kY, k = i % kY;
  int a = k >> 1, j = k & 1;
  float v = xnew[((size_t)a * kB + b) * N_X + j];
  yb[i] = v;
  ret[i] = v;
}

// ---------------------------------------------------------------------------
// Host driver
// ---------------------------------------------------------------------------
extern "C" void kernel_launch(void* const* d_in, const int* in_sizes, int n_in,
                              void* d_out, int out_size, void* d_ws, size_t ws_size,
                              hipStream_t stream) {
  (void)in_sizes; (void)n_in; (void)out_size; (void)ws_size;
  auto F = [&](int i) -> const float* { return (const float*)d_in[i]; };

  // ---- workspace bump allocator --------------------------------------------
  char* wsBase = (char*)d_ws;
  size_t off = 0;
  auto alloc  = [&](size_t bytes) -> char* {
    char* r = wsBase + off;
    off = (off + bytes + 255) & ~(size_t)255;
    return r;
  };
  auto allocF = [&](size_t n) -> float*          { return (float*)alloc(n * sizeof(float)); };
  auto allocH = [&](size_t n) -> unsigned short* { return (unsigned short*)alloc(n * 2); };

  // bf16 weights (padded)
  unsigned short* W_pm1    = allocH((size_t)kA * kH * kH);
  unsigned short* W_pm2    = allocH((size_t)kA * kH * kH);
  unsigned short* W_pmm    = allocH((size_t)kA * kZ * kH);
  unsigned short* W_pms    = allocH((size_t)kA * kZ * kH);
  unsigned short* W_dm1    = allocH((size_t)kA * kH * K_DM1);
  unsigned short* W_dm2    = allocH((size_t)kA * kM * kH);
  unsigned short* W_p1     = allocH((size_t)kA * kH * kH);
  unsigned short* W_p2     = allocH((size_t)kA * kH * kH);
  unsigned short* W_pme    = allocH((size_t)kA * kZ * kH);
  unsigned short* W_pst    = allocH((size_t)kA * kZ * kH);
  unsigned short* W_d1     = allocH((size_t)kA * kH * K_D1);
  unsigned short* W_d2     = allocH((size_t)kA * kH * kH);
  unsigned short* W_dme    = allocH((size_t)kA * N_X * kH);
  unsigned short* W_dst    = allocH((size_t)kA * N_X * kH);
  unsigned short* W_mi_ih0 = allocH((size_t)kA * 3 * kH * K_GMI);
  unsigned short* W_mi_hh0 = allocH((size_t)kA * 3 * kH * kH);
  unsigned short* W_mi_ih1 = allocH((size_t)kA * 3 * kH * kH);
  unsigned short* W_mi_hh1 = allocH((size_t)kA * 3 * kH * kH);
  unsigned short* W_ma_ih0 = allocH((size_t)3 * kH * K_GMA);
  unsigned short* W_ma_hh0 = allocH((size_t)3 * kH * kH);
  unsigned short* W_ma_ih1 = allocH((size_t)3 * kH * kH);
  unsigned short* W_ma_hh1 = allocH((size_t)3 * kH * kH);

  // f32 state + temps
  float* h_ma  = allocF((size_t)kL * kB * kH);
  float* h_mi  = allocF((size_t)kA * kL * kB * kH);
  float* yb    = allocF((size_t)kB * kY);
  float* f1    = allocF((size_t)kAB * kH);
  float* f2    = allocF((size_t)kAB * kH);
  float* muZ   = allocF((size_t)kAB * kZ);
  float* sdZ   = allocF((size_t)kAB * kZ);
  float* zma   = allocF((size_t)kAB * kZ);
  float* zmi   = allocF((size_t)kAB * kZ);
  float* logit = allocF((size_t)kAB * kM);
  float* m_t   = allocF((size_t)kAB * kM);
  float* gx    = allocF((size_t)kAB * 3 * kH);
  float* gh    = allocF((size_t)kAB * 3 * kH);
  float* muX   = allocF((size_t)kAB * N_X);
  float* sdX   = allocF((size_t)kAB * N_X);
  float* xnew  = allocF((size_t)kAB * N_X);

  // bf16 activation buffers
  unsigned short* bf_a   = allocH((size_t)kAB * kH);
  unsigned short* bf_b   = allocH((size_t)kAB * kH);
  unsigned short* bf_hma = allocH((size_t)kB * kH);
  unsigned short* in_dm1 = allocH((size_t)kAB * K_DM1);
  unsigned short* in_gma = allocH((size_t)kB * K_GMA);
  unsigned short* in_d1  = allocH((size_t)kAB * K_D1);
  unsigned short* in_gmi = allocH((size_t)kAB * K_GMI);

  auto g1 = [](long n) { return dim3((unsigned)((n + 255) / 256)); };

  auto packW = [&](const float* src, unsigned short* dst, int batch,
                   int N, int Npad, int K, int Kpad) {
    dim3 g((Kpad + 255) / 256, Npad, batch);
    k_pack_pad_bf16<<<g, 256, 0, stream>>>(src, (size_t)N * K, dst,
                                           (size_t)Npad * Kpad, N, K, Kpad);
  };
  auto packA = [&](const float* src, size_t sBS, unsigned short* dst, int batch,
                   int rows, int K, int Kpad) {
    dim3 g((Kpad + 255) / 256, rows, batch);
    k_pack_pad_bf16<<<g, 256, 0, stream>>>(src, sBS, dst,
                                           (size_t)rows * Kpad, rows, K, Kpad);
  };
  auto gemm = [&](const unsigned short* In, size_t inBS,
                  const unsigned short* Wp, size_t wBS,
                  const float* bias, size_t bBS, int Nbias,
                  float* Out, size_t oBS, int batch, int N, int Kpad, int act) {
    dim3 g((N / 16 + 7) / 8, kB / 64, batch);   // 64-row x 128-col block tiles
    k_gemm<<<g, 256, 0, stream>>>(In, inBS, Wp, wBS, bias, bBS, Nbias,
                                  Out, oBS, kB, N, Kpad, act);
  };

  // ---- quantize weights to bf16 (once per launch; deterministic) ------------
  packW(F(2),  W_pm1, kA, kH, kH, kH, kH);
  packW(F(4),  W_pm2, kA, kH, kH, kH, kH);
  packW(F(6),  W_pmm, kA, kZ, kZ, kH, kH);
  packW(F(8),  W_pms, kA, kZ, kZ, kH, kH);
  packW(F(10), W_dm1, kA, kH, kH, 586, K_DM1);
  packW(F(12), W_dm2, kA, kM, kM, kH, kH);
  packW(F(14), W_p1,  kA, kH, kH, kH, kH);
  packW(F(16), W_p2,  kA, kH, kH, kH, kH);
  packW(F(18), W_pme, kA, kZ, kZ, kH, kH);
  packW(F(20), W_pst, kA, kZ, kZ, kH, kH);
  packW(F(22), W_d1,  kA, kH, kH, 906, K_D1);
  packW(F(24), W_d2,  kA, kH, kH, kH, kH);
  packW(F(26), W_dme, kA, kX, N_X, kH, kH);
  packW(F(28), W_dst, kA, kX, N_X, kH, kH);
  packW(F(30), W_mi_ih0, kA, 3 * kH, 3 * kH, 66, K_GMI);
  packW(F(31), W_mi_hh0, kA, 3 * kH, 3 * kH, kH, kH);
  packW(F(34), W_mi_ih1, kA, 3 * kH, 3 * kH, kH, kH);
  packW(F(35), W_mi_hh1, kA, 3 * kH, 3 * kH, kH, kH);
  packW(F(38), W_ma_ih0, 1, 3 * kH, 3 * kH, K_GMA, K_GMA);
  packW(F(39), W_ma_hh0, 1, 3 * kH, 3 * kH, kH, kH);
  packW(F(42), W_ma_ih1, 1, 3 * kH, 3 * kH, kH, kH);
  packW(F(43), W_ma_hh1, 1, 3 * kH, 3 * kH, kH, kH);

  // ---- init state + outputs -------------------------------------------------
  float* retOut   = (float*)d_out;                          // (T,B,10)
  float* goalsOut = retOut + (size_t)kT * kB * kY;          // (T,B,5)
  const float* yIn = F(0);
  const float* mg  = F(1);

  k_zero<<<g1((long)kL * kB * kH), 256, 0, stream>>>(h_ma, (long)kL * kB * kH);
  k_zero<<<g1((long)kA * kL * kB * kH), 256, 0, stream>>>(h_mi, (long)kA * kL * kB * kH);
  k_copy2<<<g1(kB * kY), 256, 0, stream>>>(yIn, yb, retOut, kB * kY);

  const size_t sBH = (size_t)kB * kH;         // per-agent activation stride
  const size_t sLBH = (size_t)kL * kB * kH;   // h_mi agent stride

  for (int t = 0; t < kT - 1; ++t) {
    unsigned seed = 1234567u + 0x9E3779B9u * (unsigned)t;

    // ================= MACRO =================
    packA(h_ma + (kL - 1) * sBH, 0, bf_hma, 1, kB, kH, kH);      // hma (top layer)
    gemm(bf_hma, 0, W_pm1, (size_t)kH * kH, F(3), kH, kH, f1, sBH, kA, kH, kH, 1);
    packA(f1, sBH, bf_a, kA, kB, kH, kH);
    gemm(bf_a, sBH, W_pm2, (size_t)kH * kH, F(5), kH, kH, f2, sBH, kA, kH, kH, 1);
    packA(f2, sBH, bf_b, kA, kB, kH, kH);
    gemm(bf_b, sBH, W_pmm, (size_t)kZ * kH, F(7), kZ, kZ, muZ, (size_t)kB * kZ, kA, kZ, kH, 0);
    gemm(bf_b, sBH, W_pms, (size_t)kZ * kH, F(9), kZ, kZ, sdZ, (size_t)kB * kZ, kA, kZ, kH, 0);
    k_sample_gauss<<<g1((long)kAB * kZ), 256, 0, stream>>>(muZ, sdZ, zma, (long)kAB * kZ, seed);

    k_concat_dm1<<<g1((long)kAB * K_DM1), 256, 0, stream>>>(yb, zma, h_ma + (kL - 1) * sBH, in_dm1);
    gemm(in_dm1, (size_t)kB * K_DM1, W_dm1, (size_t)kH * K_DM1, F(11), kH, kH, f1, sBH, kA, kH, K_DM1, 1);
    packA(f1, sBH, bf_a, kA, kB, kH, kH);
    gemm(bf_a, sBH, W_dm2, (size_t)kM * kH, F(13), kM, kM, logit, (size_t)kB * kM, kA, kM, kH, 0);

    float* gOut2 = (t == kT - 2) ? (goalsOut + (size_t)(kT - 1) * kB * kA) : nullptr;
    k_macro_sample<<<g1(kAB), 256, 0, stream>>>(logit, mg + (size_t)t * kB * kA, m_t,
                                                goalsOut + (size_t)t * kB * kA, gOut2,
                                                seed + 1u);

    // gru_macro layer 0
    k_concat_gmacro<<<g1((long)kB * K_GMA), 256, 0, stream>>>(m_t, zma, in_gma);
    packA(h_ma, 0, bf_hma, 1, kB, kH, kH);
    gemm(in_gma, 0, W_ma_ih0, 0, F(40), 0, 3 * kH, gx, 0, 1, 3 * kH, K_GMA, 0);
    gemm(bf_hma, 0, W_ma_hh0, 0, F(41), 0, 3 * kH, gh, 0, 1, 3 * kH, kH, 0);
    k_gru_cell<<<dim3((kB * kH + 255) / 256, 1, 1), 256, 0, stream>>>(
        gx, gh, 0, h_ma, 0, bf_b, kB, kH);
    // gru_macro layer 1 (bf_b holds new h[0] in bf16)
    packA(h_ma + sBH, 0, bf_hma, 1, kB, kH, kH);
    gemm(bf_b, 0, W_ma_ih1, 0, F(44), 0, 3 * kH, gx, 0, 1, 3 * kH, kH, 0);
    gemm(bf_hma, 0, W_ma_hh1, 0, F(45), 0, 3 * kH, gh, 0, 1, 3 * kH, kH, 0);
    k_gru_cell<<<dim3((kB * kH + 255) / 256, 1, 1), 256, 0, stream>>>(
        gx, gh, 0, h_ma + sBH, 0, nullptr, kB, kH);

    // ================= MICRO =================
    packA(h_mi + (kL - 1) * sBH, sLBH, bf_a, kA, kB, kH, kH);    // hmi = h_mi[:, L-1]
    gemm(bf_a, sBH, W_p1, (size_t)kH * kH, F(15), kH, kH, f1, sBH, kA, kH, kH, 1);
    packA(f1, sBH, bf_b, kA, kB, kH, kH);
    gemm(bf_b, sBH, W_p2, (size_t)kH * kH, F(17), kH, kH, f2, sBH, kA, kH, kH, 1);
    packA(f2, sBH, bf_a, kA, kB, kH, kH);
    gemm(bf_a, sBH, W_pme, (size_t)kZ * kH, F(19), kZ, kZ, muZ, (size_t)kB * kZ, kA, kZ, kH, 0);
    gemm(bf_a, sBH, W_pst, (size_t)kZ * kH, F(21), kZ, kZ, sdZ, (size_t)kB * kZ, kA, kZ, kH, 0);
    k_sample_gauss<<<g1((long)kAB * kZ), 256, 0, stream>>>(muZ, sdZ, zmi, (long)kAB * kZ, seed + 2u);

    k_concat_d1<<<g1((long)kAB * K_D1), 256, 0, stream>>>(yb, m_t, zmi, h_mi, in_d1);
    gemm(in_d1, (size_t)kB * K_D1, W_d1, (size_t)kH * K_D1, F(23), kH, kH, f1, sBH, kA, kH, K_D1, 1);
    packA(f1, sBH, bf_a, kA, kB, kH, kH);
    gemm(bf_a, sBH, W_d2, (size_t)kH * kH, F(25), kH, kH, f2, sBH, kA, kH, kH, 1);
    packA(f2, sBH, bf_b, kA, kB, kH, kH);
    gemm(bf_b, sBH, W_dme, (size_t)N_X * kH, F(27), kX, kX, muX, (size_t)kB * N_X, kA, N_X, kH, 0);
    gemm(bf_b, sBH, W_dst, (size_t)N_X * kH, F(29), kX, kX, sdX, (size_t)kB * N_X, kA, N_X, kH, 0);
    k_sample_gauss<<<g1((long)kAB * N_X), 256, 0, stream>>>(muX, sdX, xnew, (long)kAB * N_X, seed + 3u);

    k_ynext<<<g1(kB * kY), 256, 0, stream>>>(xnew, yb, retOut + (size_t)(t + 1) * kB * kY);

    // gru_micro layer 0
    k_concat_gmicro<<<g1((long)kAB * K_GMI), 256, 0, stream>>>(xnew, zmi, in_gmi);
    packA(h_mi, sLBH, bf_a, kA, kB, kH, kH);                     // h_mi[:, 0]
    gemm(in_gmi, (size_t)kB * K_GMI, W_mi_ih0, (size_t)3 * kH * K_GMI, F(32), 3 * kH, 3 * kH,
         gx, (size_t)kB * 3 * kH, kA, 3 * kH, K_GMI, 0);
    gemm(bf_a, sBH, W_mi_hh0, (size_t)3 * kH * kH, F(33), 3 * kH, 3 * kH,
         gh, (size_t)kB * 3 * kH, kA, 3 * kH, kH, 0);
    k_gru_cell<<<dim3((kB * kH + 255) / 256, 1, kA), 256, 0, stream>>>(
        gx, gh, (size_t)kB * 3 * kH, h_mi, sLBH, bf_b, kB, kH);
    // gru_micro layer 1
    packA(h_mi + sBH, sLBH, bf_a, kA, kB, kH, kH);               // h_mi[:, 1]
    gemm(bf_b, sBH, W_mi_ih1, (size_t)3 * kH * kH, F(36), 3 * kH, 3 * kH,
         gx, (size_t)kB * 3 * kH, kA, 3 * kH, kH, 0);
    gemm(bf_a, sBH, W_mi_hh1, (size_t)3 * kH * kH, F(37), 3 * kH, 3 * kH,
         gh, (size_t)kB * 3 * kH, kA, 3 * kH, kH, 0);
    k_gru_cell<<<dim3((kB * kH + 255) / 256, 1, kA), 256, 0, stream>>>(
        gx, gh, (size_t)kB * 3 * kH, h_mi + sBH, sLBH, nullptr, kB, kH);
  }
}
